// Encoder_55482387530485
// MI455X (gfx1250) — matrix operand order repaired and checked by reading, not compiled, not run on hardware
//
#include <hip/hip_runtime.h>
#include <hip/hip_bf16.h>
#include <stdint.h>

#define B_   64
#define T_   256
#define D_   512
#define H_   1024
#define FH_  4096   // 4*H

typedef __attribute__((ext_vector_type(16))) __bf16          v16bf;
typedef __attribute__((ext_vector_type(8)))  float           v8f;
typedef __attribute__((ext_vector_type(8)))  unsigned short  us8;
typedef __attribute__((ext_vector_type(16))) unsigned short  us16;
typedef __attribute__((ext_vector_type(4)))  unsigned int    u32x4;
typedef __attribute__((ext_vector_type(4)))  int             i32x4;
typedef __attribute__((ext_vector_type(8)))  int             i32x8;

union Frag {
    us16  u;
    us8   h[2];
    v16bf bf;
};

__device__ __forceinline__ unsigned short f2bf(float f) {
    union { float f; unsigned u; } x; x.f = f;
    unsigned r = x.u + 0x7FFFu + ((x.u >> 16) & 1u);   // round-to-nearest-even
    return (unsigned short)(r >> 16);
}
__device__ __forceinline__ float bf2f(unsigned short s) {
    union { unsigned u; float f; } x; x.u = ((unsigned)s) << 16;
    return x.f;
}
__device__ __forceinline__ float sigm_(float x) { return 1.0f / (1.0f + __expf(-x)); }

// ---------------------------------------------------------------------------
// TDM: async-copy h (64 rows x 2048 B contiguous) from global into LDS.
// D# built per CDNA5 ISA 8.3/8.4: group0 = {count=1, lds_addr, global_addr,
// type=2}; group1 = {data_size=8B, tensor_dim0=256 (8B units),
// tensor_dim1=64, tile_dim0=256, tile_dim1=64, dim0_stride=256}.
// This toolchain's builtin takes 6 args (g0, g1, g2, g3, extra, cpol).
// ---------------------------------------------------------------------------
__device__ __forceinline__ void tdm_load_h_to_lds(const unsigned short* gsrc,
                                                  unsigned lds_off) {
    unsigned long long ga = (unsigned long long)(uintptr_t)gsrc;
    u32x4 g0;
    g0[0] = 1u;                                            // count=1 (user D#)
    g0[1] = lds_off;                                       // lds_addr (bytes)
    g0[2] = (unsigned)(ga & 0xFFFFFFFFull);                // global_addr[31:0]
    g0[3] = (unsigned)((ga >> 32) & 0x01FFFFFFull) | 0x80000000u; // [56:32]|type=2
    i32x8 g1;
    g1[0] = 0x00030000;   // workgroup_mask=0, data_size=3 (8B), no pad/iterate
    g1[1] = 0x01000000;   // tensor_dim0[15:0]=256 in bits[63:48]
    g1[2] = 0x00400000;   // tensor_dim0 hi=0 | tensor_dim1=64 in bits[95:80]
    g1[3] = 0x01000000;   // tensor_dim1 hi=0 | tile_dim0=256 in bits[127:112]
    g1[4] = 64;           // tile_dim1=64, tile_dim2=0
    g1[5] = 256;          // tensor_dim0_stride=256 (8B units)
    g1[6] = 0;
    g1[7] = 0;
    i32x4 z4 = {0, 0, 0, 0};
    i32x8 z8 = {0, 0, 0, 0, 0, 0, 0, 0};
    __builtin_amdgcn_tensor_load_to_lds(g0, g1, z4, z4, z8, 0);
}

// ---------------------------------------------------------------------------
// Pre-swizzle fp32 K x N weight into WMMA bf16 B-fragment layout.
// Tile (kt, nt): lane L owns column n = nt*16 + (L&15), 16 contiguous K
// values k = kt*32 + (L>>4)*16 + e, stored as 32 contiguous bytes per lane.
// ---------------------------------------------------------------------------
__global__ void swizzle_b(const float* __restrict__ src,
                          unsigned short* __restrict__ dst, int K, int N) {
    int total = K * N;
    int NT = N >> 4;
    for (int idx = blockIdx.x * blockDim.x + threadIdx.x; idx < total;
         idx += gridDim.x * blockDim.x) {
        int tile = idx >> 9;           // 512 elements per tile
        int r    = idx & 511;
        int lane = r >> 4, e = r & 15;
        int kt = tile / NT, nt = tile - kt * NT;
        int n = (nt << 4) + (lane & 15);
        int k = (kt << 5) + ((lane >> 4) << 4) + e;
        dst[idx] = f2bf(src[(size_t)k * N + n]);
    }
}

// Gather x = emb[tokens] -> bf16 row-major (B*T, D)
__global__ void gather_emb(const int* __restrict__ tokens,
                           const float* __restrict__ emb,
                           unsigned short* __restrict__ xb) {
    int total = B_ * T_ * D_;
    for (int idx = blockIdx.x * blockDim.x + threadIdx.x; idx < total;
         idx += gridDim.x * blockDim.x) {
        int row = idx >> 9;            // D_ = 512
        int c   = idx & 511;
        int tok = tokens[row];
        xb[idx] = f2bf(emb[(size_t)tok * D_ + c]);
    }
}

// Zero both h ping-pong buffers, build transposed mask, reset barrier counter.
__global__ void prep_misc(const int* __restrict__ tk,
                          unsigned char* __restrict__ maskT,
                          unsigned short* __restrict__ hbufs,   // 2*B*H contiguous
                          unsigned int* __restrict__ cnt) {
    int idx = blockIdx.x * blockDim.x + threadIdx.x;
    if (idx < 2 * B_ * H_) hbufs[idx] = 0;
    if (idx < B_ * T_) {
        int t = idx >> 6, b = idx & 63;
        maskT[idx] = (tk[b * T_ + t] != 0) ? 1 : 0;
    }
    if (idx == 0) *cnt = 0u;
}

// ---------------------------------------------------------------------------
// xW = x @ W + bias  -> bf16, stored (t, b, 4H): each scan step reads one
// contiguous 512 KB L2-resident slab. grid = (256 mblocks, 64 nblocks).
// ---------------------------------------------------------------------------
__global__ __launch_bounds__(128)
void xw_gemm(const unsigned short* __restrict__ Xb,   // (16384, 512) bf16
             const unsigned short* __restrict__ Wb,   // swizzled
             const float* __restrict__ bias,          // (4096)
             unsigned short* __restrict__ xWb) {      // (T, B, 4H) bf16
    __shared__ unsigned short sA[64 * 512];           // 64 KB A panel
    const int mb = blockIdx.x, nb = blockIdx.y;
    const int tid = threadIdx.x, wave = tid >> 5, lane = tid & 31;

    { // stage 64x512 bf16 A panel (64 KB) cooperatively
        const uint4* s = (const uint4*)(Xb + (size_t)mb * 64 * 512);
        uint4* d = (uint4*)sA;
        for (int i = tid; i < 4096; i += 128) d[i] = s[i];
    }
    __syncthreads();

    const int ntile = nb * 4 + wave;                  // 0..255
    const int ncol  = ntile * 16 + (lane & 15);
    v8f acc[4] = {};
#pragma unroll 4
    for (int kt = 0; kt < 16; ++kt) {
        Frag bfr;
        bfr.u = *(const us16*)(Wb + ((size_t)(kt * 256 + ntile)) * 512 + lane * 16);
        const int kb = (kt << 5) + ((lane >> 4) << 4);
        const unsigned short* base = sA + (lane & 15) * 512 + kb;
#pragma unroll
        for (int mt = 0; mt < 4; ++mt) {
            const unsigned short* hr = base + mt * 16 * 512;
            Frag af;
            af.h[0] = *(const us8*)(hr);
            af.h[1] = *(const us8*)(hr + 8);
            acc[mt] = __builtin_amdgcn_wmma_f32_16x16x32_bf16(
                false, af.bf, false, bfr.bf, (short)0, acc[mt], false, false);
        }
    }
    const float bn = bias[ncol];
#pragma unroll
    for (int mt = 0; mt < 4; ++mt)
#pragma unroll
        for (int e = 0; e < 8; ++e) {
            int m = mb * 64 + mt * 16 + e + ((lane >> 4) << 3);
            int b = m >> 8, t = m & 255;              // rows are (b*T + t)
            xWb[((size_t)(t * 64 + b)) * FH_ + ncol] = f2bf(acc[mt][e] + bn);
        }
}

// ---------------------------------------------------------------------------
// Persistent LSTM scan: 64 WGs x 128 threads (4 waves = 4 gates), WG owns a
// 16-column slice of H. LDS (272 KB): 128 KB h stage (filled by TDM each
// step) + 128 KB WG-private U tiles (loaded once) + 16 KB gate exchange.
// c/h state in registers; h ping-pongs in global with a device-scope atomic
// barrier between steps.
// ---------------------------------------------------------------------------
__global__ __launch_bounds__(128)
void lstm_scan(const unsigned short* __restrict__ Ub,     // swizzled U
               const unsigned short* __restrict__ xWb,    // (T,B,4H) bf16
               const unsigned char* __restrict__ maskT,   // (T,B)
               unsigned short* __restrict__ hb0,
               unsigned short* __restrict__ hb1,
               unsigned int* __restrict__ cnt) {
    extern __shared__ unsigned char smemRaw[];
    unsigned short* sH = (unsigned short*)smemRaw;                 // 128 KB
    unsigned short* sU = (unsigned short*)(smemRaw + 131072);      // 128 KB
    float*          sZ = (float*)(smemRaw + 262144);               // 16 KB

    const int wg = blockIdx.x;            // 0..63 -> columns n0..n0+15
    const int tid = threadIdx.x, wave = tid >> 5, lane = tid & 31;
    const int n0 = wg * 16;
    const int eb = tid >> 1;              // elementwise: batch row
    const int en = (tid & 1) * 8;         // elementwise: col sub-offset

    // hoist this WG's U tiles into LDS once: 4 waves x 32 ktiles x 1 KB
    {
        uint4* dU = (uint4*)sU;
        for (int idx = tid; idx < 8192; idx += 128) {
            int tile = idx >> 6;          // 64 uint4 per tile
            int off  = idx & 63;
            int w    = tile >> 5;
            int kt   = tile & 31;
            const uint4* s =
                (const uint4*)(Ub + ((size_t)(kt * 256 + (w * 64 + wg))) * 512);
            dU[idx] = s[off];
        }
    }

    float c_st[8], h_st[8];
#pragma unroll
    for (int j = 0; j < 8; ++j) { c_st[j] = 0.f; h_st[j] = 0.f; }

    unsigned short* bufs[2] = { hb0, hb1 };

    for (int t = 0; t < T_; ++t) {
        const unsigned short* hcur = bufs[t & 1];
        unsigned short* hnext      = bufs[(t + 1) & 1];

        // warm L2/L0 for this step's xW rows while the GEMM runs
        {
            const unsigned short* pw = xWb + ((size_t)(t * 64 + eb)) * FH_ + n0;
            __builtin_prefetch(pw, 0, 1);
            __builtin_prefetch(pw + H_, 0, 1);
            __builtin_prefetch(pw + 2 * H_, 0, 1);
            __builtin_prefetch(pw + 3 * H_, 0, 1);
        }

        // stage full h (64x1024 bf16 = 128 KB) into LDS via Tensor Data Mover
        if (wave == 0) {
            tdm_load_h_to_lds(hcur, (unsigned)(uintptr_t)sH);
            __builtin_amdgcn_s_wait_tensorcnt(0);
        }
        __syncthreads();

        // GEMM: z_gate[:, n0..n0+15] = h @ U[:, gate*H + n0 ..], all from LDS
        v8f acc[4] = {};
#pragma unroll 2
        for (int kt = 0; kt < 32; ++kt) {
            Frag bfr;
            bfr.u = *(const us16*)(sU + ((size_t)((wave * 32 + kt) * 32 + lane)) * 16);
            const int kb = (kt << 5) + ((lane >> 4) << 4);
            const unsigned short* base = sH + (lane & 15) * H_ + kb;
#pragma unroll
            for (int mt = 0; mt < 4; ++mt) {
                const unsigned short* hr = base + mt * 16 * H_;
                Frag af;
                af.h[0] = *(const us8*)(hr);
                af.h[1] = *(const us8*)(hr + 8);
                acc[mt] = __builtin_amdgcn_wmma_f32_16x16x32_bf16(
                    false, af.bf, false, bfr.bf, (short)0, acc[mt], false, false);
            }
        }
        // dump gate slice to LDS (D-fragment: row = mt*16 + e + 8*(lane>>4))
#pragma unroll
        for (int mt = 0; mt < 4; ++mt)
#pragma unroll
            for (int e = 0; e < 8; ++e) {
                int row = mt * 16 + e + ((lane >> 4) << 3);
                sZ[wave * 1024 + row * 16 + (lane & 15)] = acc[mt][e];
            }
        __syncthreads();

        // elementwise LSTM cell: thread owns (row eb, cols en..en+7)
        const unsigned short* xwrow = xWb + ((size_t)(t * 64 + eb)) * FH_ + n0;
        const bool msk = maskT[t * 64 + eb] != 0;
#pragma unroll
        for (int j = 0; j < 8; ++j) {
            int nl = en + j;
            float zi = sZ[0 * 1024 + eb * 16 + nl] + bf2f(xwrow[0 * H_ + nl]);
            float zf = sZ[1 * 1024 + eb * 16 + nl] + bf2f(xwrow[1 * H_ + nl]);
            float zg = sZ[2 * 1024 + eb * 16 + nl] + bf2f(xwrow[2 * H_ + nl]);
            float zo = sZ[3 * 1024 + eb * 16 + nl] + bf2f(xwrow[3 * H_ + nl]);
            float iv = sigm_(zi), fv = sigm_(zf);
            float gv = tanhf(zg), ov = sigm_(zo);
            float cn = fv * c_st[j] + iv * gv;
            float hn = ov * tanhf(cn);
            if (msk) { c_st[j] = cn; h_st[j] = hn; }
            hnext[eb * H_ + n0 + nl] = f2bf(h_st[j]);
        }

        // device-wide step barrier (monotonically growing target; counter is
        // reset by prep_misc every launch -> graph-replay deterministic)
        __syncthreads();
        if (tid == 0) {
            __threadfence();                          // release h writes
            atomicAdd(cnt, 1u);
            const unsigned target = 64u * (unsigned)(t + 1);
            while (__hip_atomic_load(cnt, __ATOMIC_RELAXED,
                                     __HIP_MEMORY_SCOPE_AGENT) < target)
                __builtin_amdgcn_s_sleep(8);
            __threadfence();                          // acquire others' h writes
        }
        __syncthreads();
    }
}

// ---------------------------------------------------------------------------
// mean = h@Wm + bm ; logvar = h@Wv + bv ; latent = eps*exp(0.5*lv) + mean
// 64 WGs x 128 threads: waves 0-1 -> mean (M-tiles 0-1/2-3), waves 2-3 -> lv.
// ---------------------------------------------------------------------------
__global__ __launch_bounds__(128)
void head_kernel(const unsigned short* __restrict__ hb,     // final h bf16 (64,1024)
                 const unsigned short* __restrict__ Wmb,
                 const unsigned short* __restrict__ Wvb,
                 const float* __restrict__ bm,
                 const float* __restrict__ bv,
                 const float* __restrict__ eps,
                 float* __restrict__ out) {
    __shared__ float sM[2 * 64 * 16];                       // mean | logvar
    const int nb = blockIdx.x;                              // 0..63
    const int tid = threadIdx.x, wave = tid >> 5, lane = tid & 31;
    const int which = wave >> 1;                            // 0=mean 1=logvar
    const int mhalf = wave & 1;                             // M-tile pair
    const unsigned short* Wsw = which ? Wvb : Wmb;

    v8f acc[2] = {};
#pragma unroll 2
    for (int kt = 0; kt < 32; ++kt) {
        Frag bfr;
        bfr.u = *(const us16*)(Wsw + ((size_t)(kt * 64 + nb)) * 512 + lane * 16);
        const int kb = (kt << 5) + ((lane >> 4) << 4);
#pragma unroll
        for (int mt = 0; mt < 2; ++mt) {
            int row = (mhalf * 2 + mt) * 16 + (lane & 15);
            const unsigned short* hr = hb + row * H_ + kb;
            Frag af;
            af.h[0] = *(const us8*)(hr);
            af.h[1] = *(const us8*)(hr + 8);
            acc[mt] = __builtin_amdgcn_wmma_f32_16x16x32_bf16(
                false, af.bf, false, bfr.bf, (short)0, acc[mt], false, false);
        }
    }
    const int ncol = nb * 16 + (lane & 15);
    const float bias = which ? bv[ncol] : bm[ncol];
#pragma unroll
    for (int mt = 0; mt < 2; ++mt)
#pragma unroll
        for (int e = 0; e < 8; ++e) {
            int row = (mhalf * 2 + mt) * 16 + e + ((lane >> 4) << 3);
            sM[which * 1024 + row * 16 + (lane & 15)] = acc[mt][e] + bias;
        }
    __syncthreads();

    const int eb = tid >> 1, en = (tid & 1) * 8;
#pragma unroll
    for (int j = 0; j < 8; ++j) {
        int nl = en + j;
        float mean = sM[eb * 16 + nl];
        float lv   = sM[1024 + eb * 16 + nl];
        int n = nb * 16 + nl;
        out[eb * H_ + n] = eps[eb * H_ + n] * __expf(0.5f * lv) + mean;
    }
}

// ---------------------------------------------------------------------------
extern "C" void kernel_launch(void* const* d_in, const int* in_sizes, int n_in,
                              void* d_out, int out_size, void* d_ws, size_t ws_size,
                              hipStream_t stream) {
    (void)in_sizes; (void)n_in; (void)out_size; (void)ws_size;
    const int*   tokens = (const int*)  d_in[0];
    const float* emb    = (const float*)d_in[1];
    const float* W      = (const float*)d_in[2];
    const float* U      = (const float*)d_in[3];
    const float* bias   = (const float*)d_in[4];
    const float* Wm     = (const float*)d_in[5];
    const float* bm     = (const float*)d_in[6];
    const float* Wv     = (const float*)d_in[7];
    const float* bv     = (const float*)d_in[8];
    const float* eps    = (const float*)d_in[9];
    float* out = (float*)d_out;

    unsigned char* ws = (unsigned char*)d_ws;
    size_t off = 0;
    auto alloc = [&](size_t bytes) -> void* {
        void* p = ws + off;
        off += (bytes + 255) & ~(size_t)255;
        return p;
    };
    unsigned short* Wb    = (unsigned short*)alloc((size_t)D_ * FH_ * 2);     //   4 MB
    unsigned short* Ubsw  = (unsigned short*)alloc((size_t)H_ * FH_ * 2);     //   8 MB
    unsigned short* Wmb   = (unsigned short*)alloc((size_t)H_ * H_ * 2);      //   2 MB
    unsigned short* Wvb   = (unsigned short*)alloc((size_t)H_ * H_ * 2);      //   2 MB
    unsigned short* Xb    = (unsigned short*)alloc((size_t)B_ * T_ * D_ * 2); //  16 MB
    unsigned short* xWb   = (unsigned short*)alloc((size_t)B_ * T_ * FH_ * 2);// 128 MB
    unsigned short* hb    = (unsigned short*)alloc((size_t)2 * B_ * H_ * 2);  // 256 KB
    unsigned char*  maskT = (unsigned char*) alloc((size_t)B_ * T_);
    unsigned int*   cnt   = (unsigned int*)  alloc(256);
    unsigned short* hb0 = hb;
    unsigned short* hb1 = hb + (size_t)B_ * H_;

    swizzle_b <<<2048, 256, 0, stream>>>(W,  Wb,   D_, FH_);
    swizzle_b <<<4096, 256, 0, stream>>>(U,  Ubsw, H_, FH_);
    swizzle_b <<<1024, 256, 0, stream>>>(Wm, Wmb,  H_, H_);
    swizzle_b <<<1024, 256, 0, stream>>>(Wv, Wvb,  H_, H_);
    gather_emb<<<8192, 256, 0, stream>>>(tokens, emb, Xb);
    prep_misc <<<512,  256, 0, stream>>>(tokens, maskT, hb0, cnt);

    xw_gemm<<<dim3(256, 64), 128, 0, stream>>>(Xb, Wb, bias, xWb);

    // dynamic LDS: 128 KB h stage + 128 KB U tiles + 16 KB gate exchange
    lstm_scan<<<64, 128, (size_t)(272 * 1024), stream>>>(
        Ubsw, xWb, maskT, hb0, hb1, cnt);

    // after 256 steps, final h lives in buffer 0
    head_kernel<<<64, 128, 0, stream>>>(hb0, Wmb, Wvb, bm, bv, eps, out);
}
